// SE_FBCNet_29360396435913
// MI455X (gfx1250) — matrix operand order repaired and checked by reading, not compiled, not run on hardware
//
#include <hip/hip_runtime.h>
#include <hip/hip_bf16.h>
#include <math.h>

// ---------------- problem constants ----------------
constexpr int B_   = 64;
constexpr int NB   = 9;
constexpr int NCH  = 22;
constexpr int T_   = 1000;
constexpr int G_   = 24;
constexpr int CW   = 5;
constexpr int SM   = 2;
constexpr int F_   = SM * NB * G_;     // 432
constexpr int C_   = NB * NCH;         // 198
constexpr int SF   = 4;
constexpr int NCLASS = 4;
constexpr int TCH  = T_ / SF;          // 250
constexpr float BN_EPS = 1e-5f;

// GEMM tiling
constexpr int KPAD_R = 224;            // 7 * 32  (K=198 zero-padded)
constexpr int KPAD_A = 448;            // 14 * 32 (K=432 zero-padded)
constexpr int BSTK   = 40;             // LDS B strip row pitch in bf16 (80 B, 16B aligned)
constexpr int TBLK   = 128;            // t-columns per block (4 waves x 2 tiles x 16)
constexpr int TPAD   = 1024;           // padded t rows of transposed bf16 x (zero tail)

typedef __attribute__((ext_vector_type(16))) __bf16 v16bf;
typedef __attribute__((ext_vector_type(8)))  __bf16 v8bf;
typedef __attribute__((ext_vector_type(8)))  float  v8f;
typedef __attribute__((ext_vector_type(4)))  float  v4f;

// ---- WMMA fragment loaders (layouts per CDNA5 ISA §7.12.2) ----
__device__ __forceinline__ v16bf frag_a(const __bf16* Arow, int khalf) {
  v8bf lo = *(const v8bf*)(Arow + khalf);
  v8bf hi = *(const v8bf*)(Arow + 16 + khalf);
  return __builtin_shufflevector(lo, hi, 0,1,2,3,4,5,6,7,8,9,10,11,12,13,14,15);
}
__device__ __forceinline__ v16bf frag_b(const __bf16* Brow, int kbase) {
  v8bf lo = *(const v8bf*)(Brow + kbase / 2);
  v8bf hi = *(const v8bf*)(Brow + 16 + kbase / 2);
  return __builtin_shufflevector(lo, hi, 0,1,2,3,4,5,6,7,8,9,10,11,12,13,14,15);
}

// ---- f32 staging path (used by gemm_a): branchless, clamped, masked last step ----
template <bool MASK>
__device__ __forceinline__ void stage_b(const float* __restrict__ Xn, __bf16* Bs,
                                        int ks, int t0, int Kdim, int tid) {
  constexpr int UPR = TBLK / 4;             // units per row = 32
#pragma unroll
  for (int u = 0; u < (32 * UPR) / 128; ++u) {   // 8 iterations
    int e  = u * 128 + tid;
    int kk = e / UPR;
    int c4 = (e % UPR) * 4;
    int kg = ks * 32 + kk;
    int kc = MASK ? ((kg < Kdim) ? kg : (Kdim - 1)) : kg;
    int tc = t0 + c4; tc = (tc > T_ - 4) ? (T_ - 4) : tc;
    v4f v = *(const v4f*)(Xn + (size_t)kc * T_ + tc);
    if (MASK) v *= (kg < Kdim) ? 1.f : 0.f;
    __bf16* q = &Bs[c4 * BSTK + kk];
    q[0 * BSTK] = (__bf16)v.x;
    q[1 * BSTK] = (__bf16)v.y;
    q[2 * BSTK] = (__bf16)v.z;
    q[3 * BSTK] = (__bf16)v.w;
  }
}
__device__ __forceinline__ void stage_step(const float* __restrict__ Xn, __bf16* Bs,
                                           int ks, int NK, int t0, int Kdim, int tid) {
  if (ks < NK - 1) stage_b<false>(Xn, Bs, ks, t0, Kdim, tid);
  else             stage_b<true >(Xn, Bs, ks, t0, Kdim, tid);
}

// ---- async staging path (gemm_r): xT is bf16, k-contiguous, fully zero-padded, so each
// 16-byte chunk is copied verbatim global->LDS with the async engine (ASYNCcnt). LDS VDST
// operand = low 32 bits of the flat shared-aperture address (ISA 10.2).
__device__ __forceinline__ void stage_async(const __bf16* __restrict__ XTn, __bf16* Bs,
                                            int ks, int t0, int tid) {
  const __bf16* grow = XTn + (size_t)(t0 + tid) * KPAD_R + ks * 32;   // 64B, 16B-aligned
  __bf16*       lrow = Bs + tid * BSTK;                               // pitch 80B
#pragma unroll
  for (int ch = 0; ch < 4; ++ch) {
    const __bf16* g = grow + ch * 8;
    unsigned lds = (unsigned)(unsigned long long)(const void*)(lrow + ch * 8);
    asm volatile("global_load_async_to_lds_b128 %0, %1, off"
                 :: "v"(lds), "v"(g) : "memory");
  }
}
__device__ __forceinline__ void wait_async0() {
  asm volatile("s_wait_asynccnt 0" ::: "memory");
}

// ---------------- 0) transpose x (B,C,T) f32 -> xT (B,TPAD,KPAD_R) bf16, zero-padded ----------------
__global__ __launch_bounds__(256) void txpose_kernel(const float* __restrict__ src,
                                                     __bf16* __restrict__ dst)
{
  __shared__ float tile[32][33];
  const int tx = threadIdx.x;            // 0..31
  const int ty = threadIdx.y;            // 0..7
  const int t0 = blockIdx.x * 32;
  const int k0 = blockIdx.y * 32;
  const int n  = blockIdx.z;
  const float* S = src + (size_t)n * C_ * T_;
#pragma unroll
  for (int i = 0; i < 4; ++i) {
    int k = k0 + ty + 8 * i;
    int t = t0 + tx;
    int kc = (k < C_) ? k : (C_ - 1);
    int tc = (t < T_) ? t : (T_ - 1);
    float v = S[(size_t)kc * T_ + tc];
    tile[ty + 8 * i][tx] = (k < C_ && t < T_) ? v : 0.f;
  }
  __syncthreads();
  __bf16* D = dst + (size_t)n * TPAD * KPAD_R;
#pragma unroll
  for (int i = 0; i < 4; ++i) {
    int t = t0 + ty + 8 * i;
    int k = k0 + tx;
    D[(size_t)t * KPAD_R + k] = (__bf16)tile[tx][ty + 8 * i];
  }
}

// ---------------- 1) r1/r2 GEMMs (async-staged bf16 B operand) ----------------
__global__ __launch_bounds__(128) void gemm_r_kernel(
    const __bf16* __restrict__ xT,     // (B, TPAD, KPAD_R) bf16, zero-padded
    const float* __restrict__ Wr, const float* __restrict__ br,
    const float* __restrict__ Wr2, const float* __restrict__ br2,
    float* __restrict__ r1, float* __restrict__ r2)
{
  __shared__ alignas(16) __bf16 As1[16 * KPAD_R];
  __shared__ alignas(16) __bf16 As2[16 * KPAD_R];
  __shared__ alignas(16) __bf16 Bs[2][TBLK * BSTK];   // double buffer

  const int tid  = threadIdx.x;
  const int wave = tid >> 5, lane = tid & 31;
  const int n  = blockIdx.z;
  const int f0 = blockIdx.y * 16;
  const int t0 = blockIdx.x * TBLK;
  const __bf16* XTn = xT + (size_t)n * TPAD * KPAD_R;

  {
    unsigned* a1w = (unsigned*)As1;
    unsigned* a2w = (unsigned*)As2;
    for (int e = tid; e < 16 * KPAD_R / 2; e += 128) { a1w[e] = 0u; a2w[e] = 0u; }
  }
  __syncthreads();
  for (int r = 0; r < 16; ++r) {
    const float* w1row = Wr  + (size_t)(f0 + r) * C_;
    const float* w2row = Wr2 + (size_t)(f0 + r) * C_;
    for (int c = tid; c < C_; c += 128) {
      As1[r * KPAD_R + c] = (__bf16)w1row[c];
      As2[r * KPAD_R + c] = (__bf16)w2row[c];
    }
  }
  constexpr int NK = KPAD_R / 32;              // 7
  stage_async(XTn, Bs[0], 0, t0, tid);
  wait_async0();
  __syncthreads();

  const int m     = lane & 15;
  const int khalf = (lane < 16) ? 0 : 8;
  const int kbase = (lane < 16) ? 0 : 16;
  const int tc0   = wave * 32 + (lane & 15);
  const int tc1   = tc0 + 16;
  v8f acc00 = {}, acc01 = {}, acc10 = {}, acc11 = {};

  int p = 0;
#pragma unroll 1
  for (int ks = 0; ks < NK; ++ks) {
    v16bf a1 = frag_a(&As1[m * KPAD_R + ks * 32], khalf);
    v16bf a2 = frag_a(&As2[m * KPAD_R + ks * 32], khalf);
    v16bf b0 = frag_b(&Bs[p][tc0 * BSTK], kbase);
    v16bf b1 = frag_b(&Bs[p][tc1 * BSTK], kbase);
    if (ks + 1 < NK) stage_async(XTn, Bs[1 - p], ks + 1, t0, tid);   // async engine overlaps WMMA
    acc00 = __builtin_amdgcn_wmma_f32_16x16x32_bf16(false, a1, false, b0, (short)0, acc00, false, false);
    acc01 = __builtin_amdgcn_wmma_f32_16x16x32_bf16(false, a1, false, b1, (short)0, acc01, false, false);
    acc10 = __builtin_amdgcn_wmma_f32_16x16x32_bf16(false, a2, false, b0, (short)0, acc10, false, false);
    acc11 = __builtin_amdgcn_wmma_f32_16x16x32_bf16(false, a2, false, b1, (short)0, acc11, false, false);
    wait_async0();
    __syncthreads();
    p ^= 1;
  }

  const int rbase = (lane < 16) ? 0 : 8;
#pragma unroll
  for (int half = 0; half < 2; ++half) {
    int tg = t0 + (half ? tc1 : tc0);
    if (tg < T_) {
      const v8f& a1c = half ? acc01 : acc00;
      const v8f& a2c = half ? acc11 : acc10;
#pragma unroll
      for (int r = 0; r < 8; ++r) {
        int f = f0 + rbase + r;
        size_t o = ((size_t)n * F_ + f) * T_ + tg;
        r1[o] = (a1c[r] + br [f]) * 0.1f;
        r2[o] = (a2c[r] + br2[f]) * 0.1f;
      }
    }
  }
}

// ---------------- 2) conv + BN + SiLU, added in-place onto r2 -> h_in ----------------
__global__ void xres_kernel(
    const float* __restrict__ x, const int* __restrict__ idx,
    const float* __restrict__ conv_w, const float* __restrict__ conv_b,
    const float* __restrict__ bn_g, const float* __restrict__ bn_b,
    const float* __restrict__ bn_m, const float* __restrict__ bn_v,
    float* __restrict__ hin /* r2, updated in place */)
{
  int t = blockIdx.x * blockDim.x + threadIdx.x;
  int f = blockIdx.y, n = blockIdx.z;
  if (t >= T_) return;
  int b   = f / (SM * G_);
  int rem = f % (SM * G_);
  int o   = rem / G_;
  int g   = rem % G_;

  float w[CW]; float n2 = 0.f;
#pragma unroll
  for (int c = 0; c < CW; ++c) {
    float wv = conv_w[((size_t)(g * NB + b) * SM + o) * CW + c];
    w[c] = wv; n2 += wv * wv;
  }
  float nn = sqrtf(n2);
  float wsc = (nn > 2.0f) ? (2.0f / (nn + 1e-7f)) : 1.0f;

  float acc = 0.f;
#pragma unroll
  for (int c = 0; c < CW; ++c) {
    int ic = idx[g * CW + c];
    acc += x[(((size_t)n * NB + b) * NCH + ic) * T_ + t] * (w[c] * wsc);
  }
  acc += conv_b[f];
  float bnsc = bn_g[f] * rsqrtf(bn_v[f] + BN_EPS);
  float y = (acc - bn_m[f]) * bnsc + bn_b[f];
  y = y / (1.f + expf(-y));                             // SiLU
  size_t off = ((size_t)n * F_ + f) * T_ + t;
  hin[off] = y + hin[off];                              // xres + r2
}

// ---------------- 3) h = Wa x h_in + ba + r1 (overwrites r1 in place) ----------------
__global__ __launch_bounds__(128) void gemm_a_kernel(
    const float* __restrict__ hin,    // (B, F, T)
    const float* __restrict__ Wa, const float* __restrict__ ba,
    float* __restrict__ r1h /* r1 in, h out (in place) */)
{
  __shared__ alignas(16) __bf16 As[16 * KPAD_A];
  __shared__ alignas(16) __bf16 Bs[2][TBLK * BSTK];   // double buffer

  const int tid  = threadIdx.x;
  const int wave = tid >> 5, lane = tid & 31;
  const int n  = blockIdx.z;
  const int f0 = blockIdx.y * 16;
  const int t0 = blockIdx.x * TBLK;
  const float* Xn = hin + (size_t)n * F_ * T_;

  {
    unsigned* aw = (unsigned*)As;
    for (int e = tid; e < 16 * KPAD_A / 2; e += 128) aw[e] = 0u;
  }
  __syncthreads();
  for (int r = 0; r < 16; ++r) {
    const float* wrow = Wa + (size_t)(f0 + r) * F_;
    for (int c = tid; c < F_; c += 128)
      As[r * KPAD_A + c] = (__bf16)wrow[c];
  }
  constexpr int NK = KPAD_A / 32;              // 14
  stage_step(Xn, Bs[0], 0, NK, t0, F_, tid);
  __syncthreads();

  const int m     = lane & 15;
  const int khalf = (lane < 16) ? 0 : 8;
  const int kbase = (lane < 16) ? 0 : 16;
  const int tc0   = wave * 32 + (lane & 15);
  const int tc1   = tc0 + 16;
  v8f acc0 = {}, acc1 = {};

  int p = 0;
#pragma unroll 1
  for (int ks = 0; ks < NK; ++ks) {
    v16bf a  = frag_a(&As[m * KPAD_A + ks * 32], khalf);
    v16bf b0 = frag_b(&Bs[p][tc0 * BSTK], kbase);
    v16bf b1 = frag_b(&Bs[p][tc1 * BSTK], kbase);
    if (ks + 1 < NK) stage_step(Xn, Bs[1 - p], ks + 1, NK, t0, F_, tid);
    acc0 = __builtin_amdgcn_wmma_f32_16x16x32_bf16(false, a, false, b0, (short)0, acc0, false, false);
    acc1 = __builtin_amdgcn_wmma_f32_16x16x32_bf16(false, a, false, b1, (short)0, acc1, false, false);
    __syncthreads();
    p ^= 1;
  }

  const int rbase = (lane < 16) ? 0 : 8;
#pragma unroll
  for (int half = 0; half < 2; ++half) {
    int tg = t0 + (half ? tc1 : tc0);
    if (tg < T_) {
      const v8f& ac = half ? acc1 : acc0;
#pragma unroll
      for (int r = 0; r < 8; ++r) {
        int f = f0 + rbase + r;
        size_t o = ((size_t)n * F_ + f) * T_ + tg;
        r1h[o] = ac[r] + ba[f] + r1h[o];
      }
    }
  }
}

// ---------------- 4) per-chunk variance -> lv (B,F,SF) ----------------
__global__ __launch_bounds__(256) void var_kernel(const float* __restrict__ h,
                                                  float* __restrict__ lv)
{
  int wid  = blockIdx.x * (blockDim.x >> 5) + (threadIdx.x >> 5);
  int lane = threadIdx.x & 31;
  if (wid >= B_ * F_ * SF) return;
  int s  = wid % SF;
  int nf = wid / SF;
  const float* base = h + (size_t)nf * T_ + s * TCH;
  float sum = 0.f, sq = 0.f;
  for (int i = lane; i < TCH; i += 32) { float v = base[i]; sum += v; sq += v * v; }
#pragma unroll
  for (int d = 16; d > 0; d >>= 1) {
    sum += __shfl_xor(sum, d, 32);
    sq  += __shfl_xor(sq,  d, 32);
  }
  if (lane == 0) {
    float var = (sq - sum * sum / (float)TCH) / (float)(TCH - 1);  // ddof=1
    var = fminf(fmaxf(var, 1e-6f), 1e6f);
    lv[wid] = logf(var);
  }
}

// ---------------- 5) 4x4 attention over lv -> flat (B, F*SF) ----------------
__global__ void att_kernel(const float* __restrict__ lv,
                           const float* __restrict__ Wqkv, const float* __restrict__ bqkv,
                           const float* __restrict__ Wo, const float* __restrict__ bo,
                           float* __restrict__ flat)
{
  int i = blockIdx.x * blockDim.x + threadIdx.x;
  if (i >= B_ * F_) return;
  int f = i % F_;
  float l[SF];
#pragma unroll
  for (int s = 0; s < SF; ++s) l[s] = lv[(size_t)i * SF + s];
  float wq = Wqkv[f * 3 + 0], wk = Wqkv[f * 3 + 1], wv = Wqkv[f * 3 + 2];
  float bq = bqkv[f * 3 + 0], bk = bqkv[f * 3 + 1], bv = bqkv[f * 3 + 2];
  float q[SF], k[SF], v[SF];
#pragma unroll
  for (int s = 0; s < SF; ++s) { q[s] = l[s] * wq + bq; k[s] = l[s] * wk + bk; v[s] = l[s] * wv + bv; }
#pragma unroll
  for (int a = 0; a < SF; ++a) {
    float mx = -3.402823e38f;
#pragma unroll
    for (int j = 0; j < SF; ++j) mx = fmaxf(mx, q[a] * k[j]);
    float se = 0.f, o = 0.f;
#pragma unroll
    for (int j = 0; j < SF; ++j) { float e = expf(q[a] * k[j] - mx); se += e; o += e * v[j]; }
    o /= se;
    flat[(size_t)i * SF + a] = o * Wo[f] + bo[f];
  }
}

// ---------------- 6) classifier-row renorm scales ----------------
__global__ void wlnorm_kernel(const float* __restrict__ Wl, float* __restrict__ wlscale)
{
  int k = threadIdx.x;
  if (k < NCLASS) {
    float n2 = 0.f;
    for (int j = 0; j < F_ * SF; ++j) { float w = Wl[(size_t)k * (F_ * SF) + j]; n2 += w * w; }
    float nn = sqrtf(n2);
    wlscale[k] = (nn > 0.5f) ? (0.5f / (nn + 1e-7f)) : 1.0f;
  }
}

// ---------------- 7) logits + log_softmax ----------------
__global__ __launch_bounds__(256) void final_kernel(
    const float* __restrict__ flat, const float* __restrict__ Wl,
    const float* __restrict__ bl, const float* __restrict__ wlscale,
    float* __restrict__ out)
{
  __shared__ float red[NCLASS * 256];
  int n = blockIdx.x, tid = threadIdx.x;
  float acc[NCLASS] = {0.f, 0.f, 0.f, 0.f};
  const float* fn = flat + (size_t)n * (F_ * SF);
  for (int j = tid; j < F_ * SF; j += 256) {
    float fv = fn[j];
#pragma unroll
    for (int k = 0; k < NCLASS; ++k) acc[k] += fv * Wl[(size_t)k * (F_ * SF) + j];
  }
#pragma unroll
  for (int k = 0; k < NCLASS; ++k) red[k * 256 + tid] = acc[k];
  __syncthreads();
  for (int st = 128; st > 0; st >>= 1) {
    if (tid < st) {
#pragma unroll
      for (int k = 0; k < NCLASS; ++k) red[k * 256 + tid] += red[k * 256 + tid + st];
    }
    __syncthreads();
  }
  if (tid == 0) {
    float lg[NCLASS], mx = -3.402823e38f;
#pragma unroll
    for (int k = 0; k < NCLASS; ++k) { lg[k] = red[k * 256] * wlscale[k] + bl[k]; mx = fmaxf(mx, lg[k]); }
    float se = 0.f;
#pragma unroll
    for (int k = 0; k < NCLASS; ++k) se += expf(lg[k] - mx);
    float lse = logf(se);
#pragma unroll
    for (int k = 0; k < NCLASS; ++k) out[(size_t)n * NCLASS + k] = lg[k] - mx - lse;
  }
}

// ---------------- launcher ----------------
extern "C" void kernel_launch(void* const* d_in, const int* in_sizes, int n_in,
                              void* d_out, int out_size, void* d_ws, size_t ws_size,
                              hipStream_t stream)
{
  const float* x      = (const float*)d_in[0];
  const int*   idx    = (const int*)  d_in[1];
  const float* conv_w = (const float*)d_in[2];
  const float* conv_b = (const float*)d_in[3];
  const float* bn_g   = (const float*)d_in[4];
  const float* bn_b   = (const float*)d_in[5];
  const float* bn_m   = (const float*)d_in[6];
  const float* bn_v   = (const float*)d_in[7];
  const float* Wr     = (const float*)d_in[8];
  const float* br     = (const float*)d_in[9];
  const float* Wr2    = (const float*)d_in[10];
  const float* br2    = (const float*)d_in[11];
  const float* Wa     = (const float*)d_in[12];
  const float* ba     = (const float*)d_in[13];
  const float* Wqkv   = (const float*)d_in[14];
  const float* bqkv   = (const float*)d_in[15];
  const float* Wo     = (const float*)d_in[16];
  const float* bo     = (const float*)d_in[17];
  const float* Wl     = (const float*)d_in[18];
  const float* bl     = (const float*)d_in[19];

  // workspace layout (~252 MB):
  //   [0, SZ)        r1  -> overwritten in place with h
  //   [SZ, 2SZ)      r2  -> updated in place to h_in = xres + r2
  //   [2SZ, +SXT)    xT  (bf16 transposed, zero-padded x)
  //   then lv (B*F*SF), flat (B*F*SF), wlscale (NCLASS)
  char* ws = (char*)d_ws;
  const size_t SZ  = (size_t)B_ * F_ * T_ * sizeof(float);
  const size_t SXT = (size_t)B_ * TPAD * KPAD_R * sizeof(__bf16);
  const size_t SLV = (size_t)B_ * F_ * SF * sizeof(float);
  float*  r1   = (float*)(ws);
  float*  r2   = (float*)(ws + SZ);
  __bf16* xT   = (__bf16*)(ws + 2 * SZ);
  float*  lv   = (float*)(ws + 2 * SZ + SXT);
  float*  flat = (float*)(ws + 2 * SZ + SXT + SLV);
  float*  wlsc = (float*)(ws + 2 * SZ + SXT + 2 * SLV);

  dim3 tgrid(TPAD / 32, KPAD_R / 32, B_);                // (32, 7, 64)
  txpose_kernel<<<tgrid, dim3(32, 8), 0, stream>>>(x, xT);

  dim3 gemm_grid((T_ + TBLK - 1) / TBLK, F_ / 16, B_);   // (8, 27, 64)
  gemm_r_kernel<<<gemm_grid, 128, 0, stream>>>(xT, Wr, br, Wr2, br2, r1, r2);

  dim3 xgrid((T_ + 255) / 256, F_, B_);
  xres_kernel<<<xgrid, 256, 0, stream>>>(x, idx, conv_w, conv_b, bn_g, bn_b, bn_m, bn_v, r2);

  gemm_a_kernel<<<gemm_grid, 128, 0, stream>>>(r2, Wa, ba, r1);

  int ngroups = B_ * F_ * SF;
  var_kernel<<<(ngroups + 7) / 8, 256, 0, stream>>>(r1, lv);

  att_kernel<<<(B_ * F_ + 255) / 256, 256, 0, stream>>>(lv, Wqkv, bqkv, Wo, bo, flat);

  wlnorm_kernel<<<1, 32, 0, stream>>>(Wl, wlsc);

  final_kernel<<<B_, 256, 0, stream>>>(flat, Wl, bl, wlsc, (float*)d_out);
}
